// EbbinghausModel_77068893160284
// MI455X (gfx1250) — matrix operand order repaired and checked by reading, not compiled, not run on hardware
//
#include <hip/hip_runtime.h>
#include <hip/hip_bf16.h>

typedef __attribute__((ext_vector_type(16))) __bf16 v16bf;
typedef __attribute__((ext_vector_type(8)))  float  v8f;
typedef unsigned int u32;

struct alignas(16) F4 { float v[4]; };
struct alignas(16) U4 { u32 v[4]; };
struct U4x2 { U4 a, b; };

__device__ __forceinline__ u32 pk_bf16(float a, float b) {
    unsigned short lo = __builtin_bit_cast(unsigned short, (__bf16)a);
    unsigned short hi = __builtin_bit_cast(unsigned short, (__bf16)b);
    return (u32)lo | ((u32)hi << 16);
}

// Load a 16x32 bf16 WMMA fragment (16 bf16 per lane) from two aligned 16B LDS chunks.
__device__ __forceinline__ v16bf ld_frag(const u32* p0, const u32* p1) {
    U4x2 t;
    t.a = *(const U4*)p0;
    t.b = *(const U4*)p1;
    return __builtin_bit_cast(v16bf, t);
}

// Compute per-thread decay weights wfac[m] = w[m]/count (uniform across lanes).
__device__ __forceinline__ void decay_weights(float ls, float* wv) {
    const float S = expf(ls);
    float cnt = 0.f;
#pragma unroll
    for (int m = 0; m < 8; ++m) {
        float age  = (float)(7 - m);
        float keep = (expf(-age / S) >= 0.05f) ? 1.f : 0.f;
        wv[m] = expf(-age * (age + 1.f) / (2.f * S)) * keep;
        cnt  += keep;
    }
    const float inv = 1.f / fmaxf(cnt, 1.f);
#pragma unroll
    for (int m = 0; m < 8; ++m) wv[m] *= inv;
}

// ---------------------------------------------------------------------------
// Precompute kernel: Q = embed @ W1[:, :64]^T, P = embed @ W1[:, 64:]^T
// Q,P are (66,64) f32. 17 blocks x 256 threads, one output pair per thread.
// ---------------------------------------------------------------------------
__launch_bounds__(256)
__global__ void ebb_precompute(const float* __restrict__ embed,   // (66,64)
                               const float* __restrict__ W1,      // (64,128)
                               float* __restrict__ Qp,            // (66,64)
                               float* __restrict__ Pp)            // (66,64)
{
    __shared__ float eS[66 * 64];     // 16.9 KB
    __shared__ float w1S[64 * 128];   // 32.0 KB
    const int tid = threadIdx.x;
    for (int e = tid; e < 66 * 64; e += 256)  eS[e]  = embed[e];
    for (int e = tid; e < 64 * 128; e += 256) w1S[e] = W1[e];
    __syncthreads();

    const int idx = blockIdx.x * 256 + tid;
    if (idx < 66 * 64) {
        const int t = idx >> 6, n = idx & 63;
        const float* er = eS  + t * 64;
        const float* wq = w1S + n * 128;        // W1[n, 0:64]
        const float* wp = w1S + n * 128 + 64;   // W1[n, 64:128]
        float sq = 0.f, sp = 0.f;
#pragma unroll
        for (int k = 0; k < 64; ++k) {
            const float ev = er[k];
            sq = fmaf(ev, wq[k], sq);
            sp = fmaf(ev, wp[k], sp);
        }
        Qp[idx] = sq;
        Pp[idx] = sp;
    }
}

// ---------------------------------------------------------------------------
// Main fused kernel: out1 = Q[qt] + sum_m wfac[m]*P[tok_m] + b1 (exact f32,
// computed per-lane in WMMA A-fragment order), then logits = relu(out1)@W2^T+b2
// via v_wmma_f32_16x16x32_bf16. 8 waves x 16 rows per block.
// ---------------------------------------------------------------------------
__launch_bounds__(256)
__global__ void ebb_fused(const int*   __restrict__ seqs,      // (B,48) int32
                          const int*   __restrict__ qtok,      // (B,)   int32
                          const float* __restrict__ Qp,        // (66,64)
                          const float* __restrict__ Pp,        // (66,64)
                          const float* __restrict__ b1,        // (64,)
                          const float* __restrict__ W2,        // (64,64)
                          const float* __restrict__ b2,        // (64,)
                          const float* __restrict__ lstab,     // scalar
                          float*       __restrict__ out)       // (B,64)
{
    __shared__ float QS[66 * 64];      // 16.9 KB
    __shared__ float PS[66 * 64];      // 16.9 KB
    __shared__ u32   w2S[64 * 32];     //  8.0 KB  W2 as packed bf16 pairs [n][kp]
    __shared__ float b1S[64];
    __shared__ float b2S[64];

    const int tid = threadIdx.x;
    for (int e = tid; e < 66 * 64; e += 256) { QS[e] = Qp[e]; PS[e] = Pp[e]; }
    for (int e = tid; e < 64 * 32; e += 256) {
        int n = e >> 5, kp = e & 31;
        w2S[e] = pk_bf16(W2[n * 64 + 2 * kp], W2[n * 64 + 2 * kp + 1]);
    }
    if (tid < 64)       b1S[tid]      = b1[tid];
    else if (tid < 128) b2S[tid - 64] = b2[tid - 64];
    __syncthreads();

    const int wid  = tid >> 5;
    const int lane = tid & 31;
    const int M    = lane & 15;      // A-fragment row / output column-in-tile
    const int hi   = lane >> 4;      // lane group (K-half selector)
    const int rowBase = blockIdx.x * 128 + wid * 16;
    const int row     = rowBase + M;

    float wv[8];
    decay_weights(lstab[0], wv);

    const int qt = qtok[row];
    int tk[8];
    const int sb = row * 48 + 39;                 // seqs[:, 39:47]
#pragma unroll
    for (int m = 0; m < 8; ++m) tk[m] = seqs[sb + m];

    // out1 chunks owned by this lane: dims {16t + 8*hi .. +7}, t = 0..3.
    // hv[c], c = t*8+i  <->  dim = 16t + 8*hi + i; A-frag kt elem e = hv[16kt+e].
    float hv[32];
    {
        const float* qr = QS + qt * 64;
#pragma unroll
        for (int t = 0; t < 4; ++t) {
            const int base = 16 * t + 8 * hi;
            F4 x  = *(const F4*)(qr + base);
            F4 y  = *(const F4*)(qr + base + 4);
            F4 bx = *(const F4*)(b1S + base);
            F4 by = *(const F4*)(b1S + base + 4);
#pragma unroll
            for (int i = 0; i < 4; ++i) {
                hv[t*8 + i]     = x.v[i] + bx.v[i];
                hv[t*8 + 4 + i] = y.v[i] + by.v[i];
            }
        }
    }
#pragma unroll
    for (int m = 0; m < 8; ++m) {
        const float* pr = PS + tk[m] * 64;
        const float  w  = wv[m];
#pragma unroll
        for (int t = 0; t < 4; ++t) {
            const int base = 16 * t + 8 * hi;
            F4 x = *(const F4*)(pr + base);
            F4 y = *(const F4*)(pr + base + 4);
#pragma unroll
            for (int i = 0; i < 4; ++i) {
                hv[t*8 + i]     = fmaf(w, x.v[i], hv[t*8 + i]);
                hv[t*8 + 4 + i] = fmaf(w, y.v[i], hv[t*8 + 4 + i]);
            }
        }
    }

    // ReLU + convert straight into GEMM2 A fragments (registers only).
    v16bf A2[2];
#pragma unroll
    for (int e = 0; e < 16; ++e) {
        A2[0][e] = (__bf16)fmaxf(hv[e],      0.f);
        A2[1][e] = (__bf16)fmaxf(hv[16 + e], 0.f);
    }

    // GEMM2: logits = h2 @ W2^T + b2, 16x64 @ 64x64 via 2 kt x 4 nt WMMA.
#pragma unroll
    for (int nt = 0; nt < 4; ++nt) {
        v8f c = {0.f, 0.f, 0.f, 0.f, 0.f, 0.f, 0.f, 0.f};
        const int n = nt * 16 + M;
#pragma unroll
        for (int kt = 0; kt < 2; ++kt) {
            const u32* p0 = w2S + n * 32 + kt * 16 + hi * 4;
            v16bf bfrag = ld_frag(p0, p0 + 8);
            c = __builtin_amdgcn_wmma_f32_16x16x32_bf16(
                    false, A2[kt], false, bfrag, (short)0, c, false, false);
        }
        const float bb = b2S[n];
#pragma unroll
        for (int v = 0; v < 8; ++v) {
            const int r = rowBase + v + 8 * hi;  // D-layout row -> global row
            out[r * 64 + n] = c[v] + bb;
        }
    }
}

// ---------------------------------------------------------------------------
// Fallback: round-1 fully-fused kernel (used only if d_ws is too small).
// ---------------------------------------------------------------------------
__launch_bounds__(256)
__global__ void ebb_fallback(const int*   __restrict__ seqs,
                             const int*   __restrict__ qtok,
                             const float* __restrict__ embed,
                             const float* __restrict__ W1,
                             const float* __restrict__ b1,
                             const float* __restrict__ W2,
                             const float* __restrict__ b2,
                             const float* __restrict__ lstab,
                             float*       __restrict__ out)
{
    __shared__ float embS[66 * 64];
    __shared__ u32   w1S [64 * 64];
    __shared__ u32   w2S [64 * 32];
    __shared__ float b1S[64];
    __shared__ float b2S[64];
    __shared__ u32   h2S[8 * 16 * 32];

    const int tid = threadIdx.x;
    for (int e = tid; e < 66 * 64; e += 256) embS[e] = embed[e];
    for (int e = tid; e < 64 * 64; e += 256) {
        int n = e >> 6, kp = e & 63;
        w1S[e] = pk_bf16(W1[n * 128 + 2 * kp], W1[n * 128 + 2 * kp + 1]);
    }
    for (int e = tid; e < 64 * 32; e += 256) {
        int n = e >> 5, kp = e & 31;
        w2S[e] = pk_bf16(W2[n * 64 + 2 * kp], W2[n * 64 + 2 * kp + 1]);
    }
    if (tid < 64)       b1S[tid]      = b1[tid];
    else if (tid < 128) b2S[tid - 64] = b2[tid - 64];
    __syncthreads();

    const int wid  = tid >> 5;
    const int lane = tid & 31;
    const int M    = lane & 15;
    const int hi   = lane >> 4;
    const int rowBase = blockIdx.x * 128 + wid * 16;
    const int row     = rowBase + M;

    float wv[8];
    decay_weights(lstab[0], wv);

    const int qt = qtok[row];
    int tk[8];
    const int sb = row * 48 + 39;
#pragma unroll
    for (int m = 0; m < 8; ++m) tk[m] = seqs[sb + m];

    float qv[32];
    {
        const float* er = embS + qt * 64;
#pragma unroll
        for (int t = 0; t < 4; ++t) {
            const int base = 16 * t + 8 * hi;
            F4 x = *(const F4*)(er + base);
            F4 y = *(const F4*)(er + base + 4);
#pragma unroll
            for (int i = 0; i < 4; ++i) { qv[t*8 + i] = x.v[i]; qv[t*8 + 4 + i] = y.v[i]; }
        }
    }
    float mv[32];
#pragma unroll
    for (int c = 0; c < 32; ++c) mv[c] = 0.f;
#pragma unroll
    for (int m = 0; m < 8; ++m) {
        const float* er = embS + tk[m] * 64;
        const float  w  = wv[m];
#pragma unroll
        for (int t = 0; t < 4; ++t) {
            const int base = 16 * t + 8 * hi;
            F4 x = *(const F4*)(er + base);
            F4 y = *(const F4*)(er + base + 4);
#pragma unroll
            for (int i = 0; i < 4; ++i) {
                mv[t*8 + i]     = fmaf(w, x.v[i], mv[t*8 + i]);
                mv[t*8 + 4 + i] = fmaf(w, y.v[i], mv[t*8 + 4 + i]);
            }
        }
    }

    v16bf A[4];
#pragma unroll
    for (int e = 0; e < 16; ++e) {
        A[0][e] = (__bf16)qv[e];
        A[1][e] = (__bf16)qv[16 + e];
        A[2][e] = (__bf16)mv[e];
        A[3][e] = (__bf16)mv[16 + e];
    }

    __bf16* h2b = (__bf16*)h2S;
    const int wb_bf = wid * 16 * 64;
#pragma unroll
    for (int nt = 0; nt < 4; ++nt) {
        v8f c = {0.f, 0.f, 0.f, 0.f, 0.f, 0.f, 0.f, 0.f};
        const int n = nt * 16 + M;
#pragma unroll
        for (int kt = 0; kt < 4; ++kt) {
            const u32* p0 = w1S + n * 64 + kt * 16 + hi * 4;
            v16bf bfrag = ld_frag(p0, p0 + 8);
            c = __builtin_amdgcn_wmma_f32_16x16x32_bf16(
                    false, A[kt], false, bfrag, (short)0, c, false, false);
        }
        const float bb = b1S[n];
#pragma unroll
        for (int v = 0; v < 8; ++v) {
            float r = fmaxf(c[v] + bb, 0.f);
            const int Md = v + 8 * hi;
            h2b[wb_bf + Md * 64 + n] = (__bf16)r;
        }
    }
    __syncthreads();

    const u32* h2w = h2S + wid * 512;
    v16bf G[2];
#pragma unroll
    for (int kt = 0; kt < 2; ++kt) {
        const u32* p0 = h2w + M * 32 + kt * 16 + 4 * hi;
        G[kt] = ld_frag(p0, p0 + 8);
    }
#pragma unroll
    for (int nt = 0; nt < 4; ++nt) {
        v8f c = {0.f, 0.f, 0.f, 0.f, 0.f, 0.f, 0.f, 0.f};
        const int n = nt * 16 + M;
#pragma unroll
        for (int kt = 0; kt < 2; ++kt) {
            const u32* p0 = w2S + n * 32 + kt * 16 + hi * 4;
            v16bf bfrag = ld_frag(p0, p0 + 8);
            c = __builtin_amdgcn_wmma_f32_16x16x32_bf16(
                    false, G[kt], false, bfrag, (short)0, c, false, false);
        }
        const float bb = b2S[n];
#pragma unroll
        for (int v = 0; v < 8; ++v) {
            const int r = rowBase + v + 8 * hi;
            out[r * 64 + n] = c[v] + bb;
        }
    }
}

extern "C" void kernel_launch(void* const* d_in, const int* in_sizes, int n_in,
                              void* d_out, int out_size, void* d_ws, size_t ws_size,
                              hipStream_t stream) {
    const int*   seqs  = (const int*)  d_in[0];
    const int*   qtok  = (const int*)  d_in[1];
    const float* embed = (const float*)d_in[2];
    const float* W1    = (const float*)d_in[3];
    const float* b1    = (const float*)d_in[4];
    const float* W2    = (const float*)d_in[5];
    const float* b2    = (const float*)d_in[6];
    const float* ls    = (const float*)d_in[7];

    const int B = in_sizes[1];          // query_tok count = batch
    const int grid = B / 128;           // 128 rows per block (8 waves x 16 rows)

    const size_t qp_elems = 66 * 64;
    const size_t need = 2 * qp_elems * sizeof(float);

    if (d_ws != nullptr && ws_size >= need) {
        float* Qp = (float*)d_ws;
        float* Pp = Qp + qp_elems;
        ebb_precompute<<<17, 256, 0, stream>>>(embed, W1, Qp, Pp);
        ebb_fused<<<grid, 256, 0, stream>>>(seqs, qtok, Qp, Pp, b1, W2, b2, ls,
                                            (float*)d_out);
    } else {
        ebb_fallback<<<grid, 256, 0, stream>>>(seqs, qtok, embed, W1, b1, W2, b2, ls,
                                               (float*)d_out);
    }
}
